// VectorQuantizer_46789373722999
// MI455X (gfx1250) — matrix operand order repaired and checked by reading, not compiled, not run on hardware
//
#include <hip/hip_runtime.h>
#include <hip/hip_bf16.h>

typedef __bf16 bf16;
typedef bf16  v16bf __attribute__((ext_vector_type(16)));
typedef float v8f   __attribute__((ext_vector_type(8)));

#define NB        32
#define C_DIM     256
#define HW        1024          // 32*32
#define NCODES    1024
#define NROWS     32768         // 32*32*32
#define NELEM     8388608.0f    // 32*256*32*32
#define INV_TEMP  100.0f        // 1/0.01
#define LDA       264           // 256 + 8 bf16 pad -> row stride 132 dwords, kills LDS bank conflicts

// ---------------------------------------------------------------- zero ws
__global__ void vq_zero(float* ws) {
    int i = blockIdx.x * 256 + threadIdx.x;
    if (i < 1280) ws[i] = 0.0f;
}

// ---------------------------------------------------------------- GEMM: distances
// Block = 256 threads (8 waves). Each block: 16 rows x all 1024 codes.
// d[n,k] = ||z_n||^2 + ||e_k||^2 - 2 * z_n . e_k, via bf16x3 WMMA split.
__global__ __launch_bounds__(256) void vq_gemm(const float* __restrict__ z,
                                               const float* __restrict__ emb,
                                               float* __restrict__ dist) {
    __shared__ bf16  a_hi[16 * LDA];
    __shared__ bf16  a_lo[16 * LDA];
    __shared__ float zpart[16][16];
    __shared__ float znorm[16];

    const int t        = threadIdx.x;
    const int row_base = blockIdx.x << 4;        // 16 rows per block, never crosses batch
    const int batch    = row_base >> 10;
    const int hw0      = row_base & 1023;

    // ---- load 16x256 z tile, split fp32 -> bf16 hi + bf16 residual, row norms
    {
        const int row = t & 15;
        const int cb  = t >> 4;
        const float* zp = z + (size_t)batch * (C_DIM * HW) + hw0 + row;
        float s = 0.0f;
#pragma unroll
        for (int p = 0; p < 16; ++p) {
            int c   = cb + (p << 4);
            float x = zp[(size_t)c * HW];
            bf16 h  = (bf16)x;
            bf16 l  = (bf16)(x - (float)h);
            a_hi[row * LDA + c] = h;
            a_lo[row * LDA + c] = l;
            s += x * x;
        }
        zpart[row][cb] = s;
    }
    __syncthreads();
    if (t < 16) {
        float s = 0.0f;
#pragma unroll
        for (int i = 0; i < 16; ++i) s += zpart[t][i];
        znorm[t] = s;
    }
    __syncthreads();

    const int wave = t >> 5;
    const int lane = t & 31;
    const int half = lane >> 4;      // 0: lanes 0-15, 1: lanes 16-31
    const int nloc = lane & 15;      // column within 16x16 tile
    const int m    = lane & 15;      // A-matrix row for this lane

    // each wave owns codes [wave*128, wave*128+128): 8 tiles of 16 codes
    for (int ct = 0; ct < 8; ++ct) {
        const int code = (wave << 7) + (ct << 4) + nloc;
        union { v8f v; float e[8]; } acc;
#pragma unroll
        for (int i = 0; i < 8; ++i) acc.e[i] = 0.0f;
        float ens = 0.0f;

        const float* ebase = emb + (size_t)code * C_DIM + (half << 4);

        for (int kc = 0; kc < 8; ++kc) {          // K = 256 in chunks of 32
            union { v16bf v; bf16 e[16]; } Ah, Al, Bh, Bl;

            // A fragment (16x32 bf16): lane m, element e -> c = kc*32 + half*16 + e
            const int  cb0 = (kc << 5) + (half << 4);
            const bf16* ph = &a_hi[m * LDA + cb0];
            const bf16* pl = &a_lo[m * LDA + cb0];
#pragma unroll
            for (int e2 = 0; e2 < 8; ++e2) {
                Ah.e[e2]     = ph[e2];      Al.e[e2]     = pl[e2];
                Ah.e[e2 + 8] = ph[8 + e2]; Al.e[e2 + 8] = pl[8 + e2];
            }

            // B fragment (32x16 bf16): lane col = nloc, K = half*16 + e (contiguous in emb)
            const float4* eq = (const float4*)(ebase + (kc << 5));
            float4 q0 = eq[0], q1 = eq[1], q2 = eq[2], q3 = eq[3];
            float xs[16] = { q0.x, q0.y, q0.z, q0.w,  q1.x, q1.y, q1.z, q1.w,
                             q2.x, q2.y, q2.z, q2.w,  q3.x, q3.y, q3.z, q3.w };
#pragma unroll
            for (int e2 = 0; e2 < 16; ++e2) {
                float x = xs[e2];
                bf16 h  = (bf16)x;
                bf16 l  = (bf16)(x - (float)h);
                Bh.e[e2] = h;
                Bl.e[e2] = l;
                ens += x * x;
            }

            // bf16x3 split: hi*hi + hi*lo + lo*hi  (~fp32 dot product)
            acc.v = __builtin_amdgcn_wmma_f32_16x16x32_bf16(false, Ah.v, false, Bh.v,
                                                            (short)0, acc.v, false, false);
            acc.v = __builtin_amdgcn_wmma_f32_16x16x32_bf16(false, Ah.v, false, Bl.v,
                                                            (short)0, acc.v, false, false);
            acc.v = __builtin_amdgcn_wmma_f32_16x16x32_bf16(false, Al.v, false, Bh.v,
                                                            (short)0, acc.v, false, false);
        }

        // full ||e_code||^2: this lane saw half the dims, partner lane (xor 16) the rest
        const float entot = ens + __shfl_xor(ens, 16);

        // D layout: lane holds rows half*8 + r (r = 0..7), consecutive hw -> 2x float4 stores
        const int zb0 = half << 3;
        float* dp = dist + (size_t)batch * (NCODES * HW) + (size_t)code * HW + hw0 + zb0;
        float4 o0 = make_float4(znorm[zb0 + 0] + entot - 2.0f * acc.e[0],
                                znorm[zb0 + 1] + entot - 2.0f * acc.e[1],
                                znorm[zb0 + 2] + entot - 2.0f * acc.e[2],
                                znorm[zb0 + 3] + entot - 2.0f * acc.e[3]);
        float4 o1 = make_float4(znorm[zb0 + 4] + entot - 2.0f * acc.e[4],
                                znorm[zb0 + 5] + entot - 2.0f * acc.e[5],
                                znorm[zb0 + 6] + entot - 2.0f * acc.e[6],
                                znorm[zb0 + 7] + entot - 2.0f * acc.e[7]);
        ((float4*)dp)[0] = o0;
        ((float4*)dp)[1] = o1;
    }
}

// ---------------------------------------------------------------- per-row pass
// 1024 threads = 32 waves; one wave per row. Distances are L2-resident (134MB < 192MB L2).
__global__ __launch_bounds__(1024) void vq_rows(const float* __restrict__ z,
                                                const float* __restrict__ emb,
                                                const float* __restrict__ dist,
                                                float* __restrict__ out,
                                                float* __restrict__ idxo,
                                                float* __restrict__ probs_g,
                                                float* __restrict__ scal_g) {
    __shared__ float probs_acc[NCODES];
    __shared__ float wmse[32], went[32];

    const int t    = threadIdx.x;
    const int wave = t >> 5;
    const int lane = t & 31;

    probs_acc[t] = 0.0f;
    __syncthreads();

    const int row   = blockIdx.x * 32 + wave;
    const int batch = row >> 10;
    const int hw    = row & 1023;
    const float* dr = dist + (size_t)batch * (NCODES * HW) + hw;

    // pass 1: per-lane min over k = lane + 32*i
    float dv[32];
    float dmin = 3.4e38f;
    int   kmin = 0;
#pragma unroll 4
    for (int i = 0; i < 32; ++i) {
        int k   = lane + (i << 5);
        float d = dr[(size_t)k * HW];
        dv[i] = d;
        if (d < dmin) { dmin = d; kmin = k; }
    }
    // wave argmin reduction, first-index tie-break
    for (int off = 16; off > 0; off >>= 1) {
        float d2 = __shfl_xor(dmin, off);
        int   k2 = __shfl_xor(kmin, off);
        if (d2 < dmin || (d2 == dmin && k2 < kmin)) { dmin = d2; kmin = k2; }
    }

    // pass 2: stable softmax stats, logits x = (dmin - d)/T  (<= 0)
    float ev[32];
    float s1 = 0.0f, s2 = 0.0f;
#pragma unroll 4
    for (int i = 0; i < 32; ++i) {
        float x = (dmin - dv[i]) * INV_TEMP;
        float e = __expf(x);
        ev[i] = e;
        s1 += e;
        s2 += x * e;
    }
    for (int off = 16; off > 0; off >>= 1) {
        s1 += __shfl_xor(s1, off);
        s2 += __shfl_xor(s2, off);
    }
    const float invZ = 1.0f / s1;
    const float ent  = __logf(s1) - s2 * invZ;   // -sum p*log p for this row

    // avg_probs accumulation (block-local LDS, then one global atomic per slot)
#pragma unroll 4
    for (int i = 0; i < 32; ++i)
        atomicAdd(&probs_acc[lane + (i << 5)], ev[i] * invZ);

    // gather zq, straight-through output, MSE partial
    const float* zr = z   + (size_t)batch * (C_DIM * HW) + hw;
    const float* er = emb + (size_t)kmin * C_DIM;
    float*     orow = out + (size_t)batch * (C_DIM * HW) + hw;
    float ms = 0.0f;
#pragma unroll
    for (int i = 0; i < 8; ++i) {
        int c    = lane + (i << 5);
        float q  = er[c];
        float zv = zr[(size_t)c * HW];
        float df = q - zv;
        ms += df * df;
        orow[(size_t)c * HW] = q;   // forward value of straight-through = zq
    }
    for (int off = 16; off > 0; off >>= 1) ms += __shfl_xor(ms, off);

    if (lane == 0) {
        wmse[wave] = ms;
        went[wave] = ent;
        idxo[row]  = (float)kmin;
    }
    __syncthreads();
    if (t == 0) {
        float a = 0.0f, b = 0.0f;
        for (int w = 0; w < 32; ++w) { a += wmse[w]; b += went[w]; }
        atomicAdd(&scal_g[0], a);
        atomicAdd(&scal_g[1], b);
    }
    atomicAdd(&probs_g[t], probs_acc[t]);
}

// ---------------------------------------------------------------- finalize losses
__global__ __launch_bounds__(1024) void vq_final(const float* __restrict__ probs_g,
                                                 const float* __restrict__ scal_g,
                                                 float* __restrict__ o) {
    __shared__ float red[1024];
    const int t = threadIdx.x;
    float ap = probs_g[t] * (1.0f / (float)NROWS);
    red[t] = -ap * __logf(ap + 1e-5f);
    __syncthreads();
    for (int s = 512; s > 0; s >>= 1) {
        if (t < s) red[t] += red[t + s];
        __syncthreads();
    }
    if (t == 0) {
        float avg_ent  = red[0];
        float mse_mean = scal_g[0] * (1.0f / NELEM);
        float commit   = 0.25f * mse_mean;     // commitment_loss
        float codebook = mse_mean;             // codebook_loss (same forward value /0.25)
        float sent     = scal_g[1] * (1.0f / (float)NROWS);
        float entloss  = 0.1f * (sent - avg_ent);
        o[0] = commit + codebook + entloss;    // loss
        o[1] = commit;
        o[2] = codebook;
        o[3] = entloss;
    }
}

// ---------------------------------------------------------------- launch
extern "C" void kernel_launch(void* const* d_in, const int* in_sizes, int n_in,
                              void* d_out, int out_size, void* d_ws, size_t ws_size,
                              hipStream_t stream) {
    const float* z   = (const float*)d_in[0];   // (32, 256, 32, 32)
    const float* emb = (const float*)d_in[1];   // (1024, 256)

    float* out   = (float*)d_out;               // (32,256,32,32)       @ 0
    float* scal  = out + 8388608;               // 4 loss scalars       @ 8388608
    float* idx   = out + 8388612;               // (32,32,32) as float  @ 8388612
    float* dist  = out + 8421380;               // (32,1024,32,32)      @ 8421380

    float* ws = (float*)d_ws;                   // [0..1023]=avg_probs sum, [1024]=mse, [1025]=ent

    vq_zero <<<5, 256, 0, stream>>>(ws);
    vq_gemm <<<NROWS / 16, 256, 0, stream>>>(z, emb, dist);
    vq_rows <<<NROWS / 32, 1024, 0, stream>>>(z, emb, dist, out, idx, ws, ws + 1024);
    vq_final<<<1, 1024, 0, stream>>>(ws, ws + 1024, scal);
}
